// RoutingLayer_51196010168971
// MI455X (gfx1250) — matrix operand order repaired and checked by reading, not compiled, not run on hardware
//
#include <hip/hip_runtime.h>

// ---------------------------------------------------------------------------
// Sinkhorn attention for MI455X (gfx1250, wave32, WMMA + TDM).
// B=2, L=1024, E=1024, H=8, D=64, N_ITERS=20.
//
//  * logits (64MB fp32) resides in the 192MB L2; Sinkhorn passes only READ it
//    and update 64KB row/col potentials r,c  (la = logits - r - c).
//  * GEMMs: v_wmma_f32_16x16x32_bf16, LDS-staged operands, fragments built
//    with 2x ds_load_b128 each. A-tiles of the big GEMMs are staged by the
//    Tensor Data Mover (tensor_load_to_lds, double-buffered, s_wait_tensorcnt).
//  * attn = exp(logits - r - c) built per-tile in registers inside attn@v.
// ---------------------------------------------------------------------------

typedef __bf16 bf16_t;
typedef bf16_t v16bf __attribute__((ext_vector_type(16)));
typedef float  v8f   __attribute__((ext_vector_type(8)));
typedef unsigned int u32x4 __attribute__((ext_vector_type(4)));
typedef int          i32x8 __attribute__((ext_vector_type(8)));
typedef int          i32x4 __attribute__((ext_vector_type(4)));

#define Lq    1024
#define Edim  1024
#define HD    512
#define Dh    64
#define Mrows 2048   // B*L
#define BH    16     // B*H
#define ASTRIDE 40   // padded LDS row stride in bf16 (80B, 16B-aligned)
#define QSTRIDE 72   // padded stride for 64-wide tiles (144B, 16B-aligned)
#define VSTRIDE 1032 // padded stride for 1024-wide V slice (2064B, 16B-aligned)

// fp32 -> bf16 (round to nearest even), raw bits
__device__ __forceinline__ unsigned short bf_bits(float f) {
    union { float f; unsigned u; } v; v.f = f;
    unsigned u = v.u + 0x7FFFu + ((v.u >> 16) & 1u);
    return (unsigned short)(u >> 16);
}

__device__ __forceinline__ v8f wmma_bf16(v16bf a, v16bf b, v8f c) {
    return __builtin_amdgcn_wmma_f32_16x16x32_bf16(
        false, a, false, b, (short)0, c, false, false);
}

union FragBF {
    v16bf v;
    unsigned u[8];
    unsigned short s[16];
};

// --- fragment loads from LDS (ISA 7.12.2 layouts), 2x ds_load_b128 each ----
// A tile in LDS: row-major [128][ASTRIDE]; lane row=lane&15 (+wave*16),
// K elements {khalf..khalf+7, khalf+16..khalf+23}.
__device__ __forceinline__ void frag_a_lds(FragBF& fa, const unsigned short* As,
                                           int rowbase, int stride, int kcol) {
    int lane  = threadIdx.x & 31;
    int row   = lane & 15;
    int khalf = (lane >> 4) * 8;
    const unsigned short* p = As + (rowbase + row) * stride + kcol + khalf;
    *(uint4*)&fa.u[0] = *(const uint4*)p;
    *(uint4*)&fa.u[4] = *(const uint4*)(p + 16);
}
// B tile in LDS transposed: [n][k] rows of stride elements; lane col=lane&15,
// K elements {kb..kb+7, kb+16..kb+23}.
__device__ __forceinline__ void frag_b_lds(FragBF& fb, const unsigned short* Bs,
                                           int stride, int kcol) {
    int lane = threadIdx.x & 31;
    int col  = lane & 15;
    int kb   = (lane >> 4) * 8;
    const unsigned short* p = Bs + col * stride + kcol + kb;
    *(uint4*)&fb.u[0] = *(const uint4*)p;
    *(uint4*)&fb.u[4] = *(const uint4*)(p + 16);
}

// --- TDM: async 2D tile load (32 x 128 bf16) global -> LDS, padded rows -----
// pad: interval = 16 DWORDs (one 32-elem row), amount = 4 DWORDs (8 halfwords)
// -> LDS row stride = ASTRIDE = 40 halfwords.
__device__ __forceinline__ void tdm_load_a_tile(const unsigned short* Ag, int lda,
                                                int m0, int k, unsigned short* As) {
    unsigned long long gaddr =
        (unsigned long long)(const void*)(Ag + (size_t)m0 * lda + k);
    unsigned lds = (unsigned)(unsigned long long)(void*)As;
    u32x4 g0;
    g0[0] = 1u;                                   // count=1
    g0[1] = lds;                                  // lds_addr
    g0[2] = (unsigned)gaddr;                      // global_addr lo
    g0[3] = (unsigned)(gaddr >> 32) | (2u << 30); // global_addr hi | type=2
    i32x8 g1;
    g1[0] = (1 << 16)            // data_size = 2 bytes
          | (1 << 20)            // pad_enable
          | (3 << 22)            // pad_interval = 16 DWORDs
          | (3 << 25);           // pad_amount   = 4 DWORDs
    g1[1] = 0;                   // tensor_dim0[15:0]=0 (dim0=0x40000000)
    g1[2] = 0x4000;              // tensor_dim0[31:16] | tensor_dim1[15:0]
    g1[3] = 0x4000 | (32 << 16); // tensor_dim1[31:16] | tile_dim0 = 32
    g1[4] = 128;                 // tile_dim1 = 128, tile_dim2 = 0
    g1[5] = lda;                 // tensor_dim0_stride lo (elements)
    g1[6] = 0;
    g1[7] = 0;
    i32x4 z4 = {};
    i32x8 z8 = {};
    __builtin_amdgcn_tensor_load_to_lds(g0, g1, z4, z4, z8, 0);
}

// --- cooperative B-tile stage: global KxN (row-major) -> LDS [n][k] ---------
__device__ __forceinline__ void stage_b_tile(unsigned short* Bs,
                                             const unsigned short* Bg,
                                             int ldb, int k, int n0) {
    int t  = threadIdx.x;
    int n  = t & 15;
    int kk = t >> 4;  // 0..15
    Bs[n * ASTRIDE + kk]      = Bg[(size_t)(k + kk) * ldb + n0 + n];
    Bs[n * ASTRIDE + kk + 16] = Bg[(size_t)(k + kk + 16) * ldb + n0 + n];
}

// --- shared GEMM mainloop: block = 8 waves = 128(M) x 16(N), TDM A staging --
template <int KDIM>
__device__ __forceinline__ v8f gemm_mainloop(const unsigned short* Ag, int lda,
                                             const unsigned short* Bg, int ldb,
                                             int m0blk, int n0,
                                             unsigned short* As, unsigned short* Bs) {
    const int NS = KDIM / 32;
    int wave = threadIdx.x >> 5;
    v8f acc = {};
    FragBF fa, fb;
    // prologue: fill buffer 0
    stage_b_tile(Bs, Bg, ldb, 0, n0);
    if (wave == 0) {
        tdm_load_a_tile(Ag, lda, m0blk, 0, As);
        __builtin_amdgcn_s_wait_tensorcnt(0);
    }
    __syncthreads();
    for (int s = 0; s < NS; ++s) {
        int cur = s & 1, nxt = cur ^ 1;
        if (s + 1 < NS) {   // prefetch next tiles into other buffer
            stage_b_tile(Bs + nxt * (16 * ASTRIDE), Bg, ldb, (s + 1) * 32, n0);
            if (wave == 0)
                tdm_load_a_tile(Ag, lda, m0blk, (s + 1) * 32,
                                As + nxt * (128 * ASTRIDE));
        }
        frag_a_lds(fa, As + cur * (128 * ASTRIDE), wave * 16, ASTRIDE, 0);
        frag_b_lds(fb, Bs + cur * (16 * ASTRIDE), ASTRIDE, 0);
        acc = wmma_bf16(fa.v, fb.v, acc);
        if (s + 1 < NS) {
            if (wave == 0) __builtin_amdgcn_s_wait_tensorcnt(0);
            __syncthreads();
        }
    }
    return acc;
}

// ---------------------------------------------------------------------------
__global__ void cvt_f32_bf16_kernel(const float* __restrict__ in,
                                    unsigned short* __restrict__ out, int n) {
    int i = blockIdx.x * blockDim.x + threadIdx.x;
    if (i < n) out[i] = bf_bits(in[i]);
}

__global__ void zero_vec_kernel(float* p, int n) {
    int i = blockIdx.x * blockDim.x + threadIdx.x;
    if (i < n) p[i] = 0.0f;
}

// ---------------------------------------------------------------------------
// QKV projection: out_bf[2048x512] = bf16( x_bf @ W_bf + bias ). grid=(32,16)
__global__ void __launch_bounds__(256)
proj_gemm_kernel(const unsigned short* __restrict__ xbf,
                 const unsigned short* __restrict__ wbf,
                 const float* __restrict__ bias,
                 unsigned short* __restrict__ outbf) {
    __shared__ __align__(16) unsigned short As[2 * 128 * ASTRIDE];
    __shared__ __align__(16) unsigned short Bs[2 * 16 * ASTRIDE];
    int m0 = blockIdx.y * 128;
    int n0 = blockIdx.x * 16;
    v8f acc = gemm_mainloop<Edim>(xbf, Edim, wbf, HD, m0, n0, As, Bs);
    int lane = threadIdx.x & 31;
    int wave = threadIdx.x >> 5;
    int col  = lane & 15;
    int roff = (lane >> 4) * 8;
    float bn = bias[n0 + col];
#pragma unroll
    for (int r = 0; r < 8; ++r)
        outbf[(size_t)(m0 + wave * 16 + roff + r) * HD + n0 + col] =
            bf_bits(acc[r] + bn);
}

// Output projection: out_f32[2048x1024] = attn_bf @ Wo_bf + bo. grid=(64,16)
__global__ void __launch_bounds__(256)
out_gemm_kernel(const unsigned short* __restrict__ abf,
                const unsigned short* __restrict__ wobf,
                const float* __restrict__ bo,
                float* __restrict__ out) {
    __shared__ __align__(16) unsigned short As[2 * 128 * ASTRIDE];
    __shared__ __align__(16) unsigned short Bs[2 * 16 * ASTRIDE];
    int m0 = blockIdx.y * 128;
    int n0 = blockIdx.x * 16;
    v8f acc = gemm_mainloop<HD>(abf, HD, wobf, Edim, m0, n0, As, Bs);
    int lane = threadIdx.x & 31;
    int wave = threadIdx.x >> 5;
    int col  = lane & 15;
    int roff = (lane >> 4) * 8;
    float bn = bo[n0 + col];
#pragma unroll
    for (int r = 0; r < 8; ++r)
        out[(size_t)(m0 + wave * 16 + roff + r) * Edim + n0 + col] = acc[r] + bn;
}

// ---------------------------------------------------------------------------
// logits[bh][q][k] = (Q . K) / 8. Whole K=64 staged once. grid=(64,8,16)
__global__ void __launch_bounds__(256)
logits_gemm_kernel(const unsigned short* __restrict__ qbf,
                   const unsigned short* __restrict__ kbf,
                   float* __restrict__ logits) {
    __shared__ __align__(16) unsigned short Qs[128 * QSTRIDE];
    __shared__ __align__(16) unsigned short Ks[16 * QSTRIDE];
    int t = threadIdx.x;
    int wave = t >> 5, lane = t & 31;
    int bh = blockIdx.z, b = bh >> 3, h = bh & 7;
    int q0blk = blockIdx.y * 128;
    int k0 = blockIdx.x * 16;
    const unsigned short* Qp = qbf + (size_t)b * Lq * HD + h * Dh;
    const unsigned short* Kp = kbf + (size_t)b * Lq * HD + h * Dh;
    {   // stage Q tile 128x64: 2 threads/row, 4x b128 each
        int row = t >> 1, half = t & 1;
        const uint4* src = (const uint4*)(Qp + (size_t)(q0blk + row) * HD + half * 32);
        uint4* dst = (uint4*)(Qs + row * QSTRIDE + half * 32);
#pragma unroll
        for (int j = 0; j < 4; ++j) dst[j] = src[j];
    }
    {   // stage K tile 16x64 -> Ks[n][d] (transposed-for-B source rows)
        int row = t >> 4, part = t & 15;
        *(uint2*)(Ks + row * QSTRIDE + part * 4) =
            *(const uint2*)(Kp + (size_t)(k0 + row) * HD + part * 4);
    }
    __syncthreads();
    v8f acc = {};
    FragBF fa, fb;
#pragma unroll
    for (int d = 0; d < Dh; d += 32) {
        frag_a_lds(fa, Qs, wave * 16, QSTRIDE, d);
        frag_b_lds(fb, Ks, QSTRIDE, d);
        acc = wmma_bf16(fa.v, fb.v, acc);
    }
    float* Lp = logits + (size_t)bh * Lq * Lq;
    int col = lane & 15;
    int roff = (lane >> 4) * 8;
#pragma unroll
    for (int r = 0; r < 8; ++r)
        Lp[(size_t)(q0blk + wave * 16 + roff + r) * Lq + k0 + col] = acc[r] * 0.125f;
}

// ---------------------------------------------------------------------------
// Row pass: r[bh][i] = logsumexp_j( logits[bh][i][j] - c[bh][j] ). grid=(1024,16)
__global__ void __launch_bounds__(256)
sinkhorn_row_kernel(const float* __restrict__ logits,
                    const float* __restrict__ c,
                    float* __restrict__ r) {
    int bh = blockIdx.y, row = blockIdx.x, t = threadIdx.x;
    const float4* Lp4 = (const float4*)(logits + ((size_t)bh * Lq + row) * Lq);
    const float4* cp4 = (const float4*)(c + bh * Lq);
    float4 xv = Lp4[t], cv = cp4[t];
    float x0 = xv.x - cv.x, x1 = xv.y - cv.y, x2 = xv.z - cv.z, x3 = xv.w - cv.w;
    float m = fmaxf(fmaxf(x0, x1), fmaxf(x2, x3));
    __shared__ float sred[256];
    sred[t] = m; __syncthreads();
    for (int s = 128; s > 0; s >>= 1) {
        if (t < s) sred[t] = fmaxf(sred[t], sred[t + s]);
        __syncthreads();
    }
    m = sred[0]; __syncthreads();
    float sum = __expf(x0 - m) + __expf(x1 - m) + __expf(x2 - m) + __expf(x3 - m);
    sred[t] = sum; __syncthreads();
    for (int s = 128; s > 0; s >>= 1) {
        if (t < s) sred[t] += sred[t + s];
        __syncthreads();
    }
    if (t == 0) r[bh * Lq + row] = m + __logf(sred[0]);
}

// Col pass stage 1: partial (max, sumexp) over 128-row chunks. grid=(4,8,16)
__global__ void __launch_bounds__(256)
sinkhorn_col_partial_kernel(const float* __restrict__ logits,
                            const float* __restrict__ r,
                            float* __restrict__ pm, float* __restrict__ ps) {
    int bh = blockIdx.z, chunk = blockIdx.y;
    int j = blockIdx.x * 256 + threadIdx.x;
    const float* Lp = logits + (size_t)bh * Lq * Lq;
    const float* rp = r + bh * Lq;
    int i0 = chunk * 128;
    float m = -3.0e38f;
    for (int i = i0; i < i0 + 128; ++i)
        m = fmaxf(m, Lp[(size_t)i * Lq + j] - rp[i]);
    float s = 0.0f;
    for (int i = i0; i < i0 + 128; ++i)
        s += __expf(Lp[(size_t)i * Lq + j] - rp[i] - m);
    size_t o = ((size_t)bh * 8 + chunk) * Lq + j;
    pm[o] = m; ps[o] = s;
}

// Col pass stage 2: combine 8 partials. grid=(64)
__global__ void __launch_bounds__(256)
sinkhorn_col_combine_kernel(const float* __restrict__ pm,
                            const float* __restrict__ ps,
                            float* __restrict__ c) {
    int idx = blockIdx.x * 256 + threadIdx.x;  // over BH*Lq
    int bh = idx >> 10, j = idx & (Lq - 1);
    float M = -3.0e38f;
#pragma unroll
    for (int ch = 0; ch < 8; ++ch)
        M = fmaxf(M, pm[((size_t)bh * 8 + ch) * Lq + j]);
    float S = 0.0f;
#pragma unroll
    for (int ch = 0; ch < 8; ++ch) {
        size_t o = ((size_t)bh * 8 + ch) * Lq + j;
        S += ps[o] * __expf(pm[o] - M);
    }
    c[idx] = M + __logf(S);
}

// ---------------------------------------------------------------------------
// attn@v: out[bh][q][d] = sum_k exp(logits - r - c) V[k][d].
// V column-slice (1024x16) + c cached in LDS once; no barriers in k-loop.
// grid = (4 d-tiles, 8 q-blocks, 16 bh)
__global__ void __launch_bounds__(256)
attnv_gemm_kernel(const float* __restrict__ logits,
                  const float* __restrict__ r,
                  const float* __restrict__ c,
                  const unsigned short* __restrict__ vbf,
                  unsigned short* __restrict__ abf) {
    __shared__ __align__(16) unsigned short Vs[16 * VSTRIDE];
    __shared__ float cs[Lq];
    int t = threadIdx.x;
    int wave = t >> 5, lane = t & 31;
    int bh = blockIdx.z, b = bh >> 3, h = bh & 7;
    int q0 = blockIdx.y * 128 + wave * 16;
    int d0 = blockIdx.x * 16;
    const float* Lp = logits + (size_t)bh * Lq * Lq;
    const float* rp = r + bh * Lq;
    const float* cp = c + bh * Lq;
    const unsigned short* Vp = vbf + (size_t)b * Lq * HD + h * Dh;
    // stage V slice: coalesced b128 reads, transposed u16 LDS stores
#pragma unroll
    for (int j = 0; j < 4; ++j) {
        int k = t + 256 * j;
        union { uint4 q; unsigned short s[8]; } ua, ub;
        const uint4* src = (const uint4*)(Vp + (size_t)k * HD + d0);
        ua.q = src[0]; ub.q = src[1];
#pragma unroll
        for (int n = 0; n < 8; ++n) {
            Vs[n * VSTRIDE + k] = ua.s[n];
            Vs[(n + 8) * VSTRIDE + k] = ub.s[n];
        }
        cs[k] = cp[k];
    }
    __syncthreads();
    int row   = lane & 15;
    int khalf = (lane >> 4) * 8;
    float rrow = rp[q0 + row];
    v8f acc = {};
    FragBF fa, fb;
    for (int k = 0; k < Lq; k += 32) {
        const float* lrow = Lp + (size_t)(q0 + row) * Lq + k + khalf;
        const float* crow = cs + k + khalf;
#pragma unroll
        for (int pi = 0; pi < 8; ++pi) {
            int kb = (pi & 3) * 2 + ((pi >> 2) * 16);
            float2 lx = *(const float2*)(lrow + kb);
            float2 cx = *(const float2*)(crow + kb);
            fa.s[2 * pi]     = bf_bits(__expf(lx.x - rrow - cx.x));
            fa.s[2 * pi + 1] = bf_bits(__expf(lx.y - rrow - cx.y));
        }
        frag_b_lds(fb, Vs, VSTRIDE, k);
        acc = wmma_bf16(fa.v, fb.v, acc);
    }
    int col = lane & 15;
    int roff = (lane >> 4) * 8;
#pragma unroll
    for (int rr = 0; rr < 8; ++rr) {
        int q = q0 + roff + rr;
        abf[((size_t)b * Lq + q) * HD + h * Dh + d0 + col] = bf_bits(acc[rr]);
    }
}

// ---------------------------------------------------------------------------
extern "C" void kernel_launch(void* const* d_in, const int* in_sizes, int n_in,
                              void* d_out, int out_size, void* d_ws, size_t ws_size,
                              hipStream_t stream) {
    const float* x  = (const float*)d_in[0];
    const float* Wq = (const float*)d_in[1];
    const float* bq = (const float*)d_in[2];
    const float* Wk = (const float*)d_in[3];
    const float* bk = (const float*)d_in[4];
    const float* Wv = (const float*)d_in[5];
    const float* bv = (const float*)d_in[6];
    const float* Wo = (const float*)d_in[7];
    const float* bo = (const float*)d_in[8];
    float* out = (float*)d_out;

    // ---- workspace carve ----
    char* ws = (char*)d_ws;
    unsigned short* xbf  = (unsigned short*)ws; ws += (size_t)Mrows * Edim * 2;
    unsigned short* wqbf = (unsigned short*)ws; ws += (size_t)Edim * HD * 2;
    unsigned short* wkbf = (unsigned short*)ws; ws += (size_t)Edim * HD * 2;
    unsigned short* wvbf = (unsigned short*)ws; ws += (size_t)Edim * HD * 2;
    unsigned short* wobf = (unsigned short*)ws; ws += (size_t)HD * Edim * 2;
    unsigned short* qbf  = (unsigned short*)ws; ws += (size_t)Mrows * HD * 2;
    unsigned short* kbf  = (unsigned short*)ws; ws += (size_t)Mrows * HD * 2;
    unsigned short* vbf  = (unsigned short*)ws; ws += (size_t)Mrows * HD * 2;
    unsigned short* abf  = (unsigned short*)ws; ws += (size_t)Mrows * HD * 2;
    float* rvec = (float*)ws; ws += (size_t)BH * Lq * 4;
    float* cvec = (float*)ws; ws += (size_t)BH * Lq * 4;
    float* pm   = (float*)ws; ws += (size_t)BH * 8 * Lq * 4;
    float* psum = (float*)ws; ws += (size_t)BH * 8 * Lq * 4;
    float* logits = (float*)ws; ws += (size_t)BH * Lq * Lq * 4;   // 64 MB

    // ---- 1. fp32 -> bf16 ----
    auto cvt = [&](const float* src, unsigned short* dst, int n) {
        cvt_f32_bf16_kernel<<<(n + 255) / 256, 256, 0, stream>>>(src, dst, n);
    };
    cvt(x,  xbf,  Mrows * Edim);
    cvt(Wq, wqbf, Edim * HD);
    cvt(Wk, wkbf, Edim * HD);
    cvt(Wv, wvbf, Edim * HD);
    cvt(Wo, wobf, HD * Edim);

    // ---- 2. QKV projections (TDM-staged WMMA) ----
    dim3 pgrid(HD / 16, Mrows / 128);
    proj_gemm_kernel<<<pgrid, 256, 0, stream>>>(xbf, wqbf, bq, qbf);
    proj_gemm_kernel<<<pgrid, 256, 0, stream>>>(xbf, wkbf, bk, kbf);
    proj_gemm_kernel<<<pgrid, 256, 0, stream>>>(xbf, wvbf, bv, vbf);

    // ---- 3. logits = q k^T / 8 ----
    logits_gemm_kernel<<<dim3(Lq / 16, Lq / 128, BH), 256, 0, stream>>>(qbf, kbf, logits);

    // ---- 4. Sinkhorn: zero potentials, 20 alternating row/col passes ----
    zero_vec_kernel<<<(BH * Lq + 255) / 256, 256, 0, stream>>>(rvec, BH * Lq);
    zero_vec_kernel<<<(BH * Lq + 255) / 256, 256, 0, stream>>>(cvec, BH * Lq);
    for (int it = 0; it < 20; ++it) {
        sinkhorn_row_kernel<<<dim3(Lq, BH), 256, 0, stream>>>(logits, cvec, rvec);
        sinkhorn_col_partial_kernel<<<dim3(Lq / 256, 8, BH), 256, 0, stream>>>(
            logits, rvec, pm, psum);
        sinkhorn_col_combine_kernel<<<BH * Lq / 256, 256, 0, stream>>>(pm, psum, cvec);
    }

    // ---- 5. out = exp(logits - r - c) @ v ----
    attnv_gemm_kernel<<<dim3(Dh / 16, Lq / 128, BH), 256, 0, stream>>>(
        logits, rvec, cvec, vbf, abf);

    // ---- 6. output projection ----
    out_gemm_kernel<<<dim3(Edim / 16, Mrows / 128), 256, 0, stream>>>(abf, wobf, bo, out);
}
